// FloorplanModel_8778913153108
// MI455X (gfx1250) — compile-verified, matrix operand order checked by reading, not run on hardware
//
#include <hip/hip_runtime.h>
#include <hip/hip_bf16.h>

// ---------------------------------------------------------------------------
// Types for CDNA5 WMMA
// ---------------------------------------------------------------------------
typedef __attribute__((ext_vector_type(16))) __bf16 v16bf;
typedef __attribute__((ext_vector_type(8)))  float  v8f;

struct FragBits { unsigned int u[8]; };

__device__ __forceinline__ unsigned short f2bf(float f) {
  unsigned int u = __float_as_uint(f);
  u += 0x7FFFu + ((u >> 16) & 1u);          // round-to-nearest-even
  return (unsigned short)(u >> 16);
}

// Pack two fp32 into one dword of bf16x2; lowers to v_cvt_pk_bf16_f32.
__device__ __forceinline__ unsigned int pack_bf16(float fx, float fy) {
  float2 t; t.x = fx; t.y = fy;
  __hip_bfloat162 h = __float22bfloat162_rn(t);
  unsigned int u;
  __builtin_memcpy(&u, &h, 4);
  return u;
}

__device__ __forceinline__ v8f wmma_bf16(v16bf a, v16bf b, v8f c) {
  return __builtin_amdgcn_wmma_f32_16x16x32_bf16(false, a, false, b, (short)0, c, false, false);
}

__device__ __forceinline__ float apply_act(float v, int act) {
  if (act == 1) return fmaxf(v, 0.f);
  if (act == 2) return 1.0f / (1.0f + __expf(-v));
  return v;
}

// ---------------------------------------------------------------------------
// Weight pack: fp32 -> bf16 (row-major [K][Cout], i.e. flattened HWIO)
// ---------------------------------------------------------------------------
__global__ void w_to_bf16(const float* __restrict__ w, unsigned short* __restrict__ o, int n) {
  int i = blockIdx.x * blockDim.x + threadIdx.x;
  if (i < n) o[i] = f2bf(w[i]);
}

// ---------------------------------------------------------------------------
// FAST implicit-GEMM conv: stride 1, SAME, odd kernel, REQUIRES Cin % 32 == 0.
//   Block: 256 threads = 8 waves.  Block tile: 128(M) x 64(N).
//   Each wave owns a 32x32 output tile -> 4 v_wmma_f32_16x16x32_bf16 / chunk.
//   Software-pipelined: double-buffered LDS, branchless (clamped) global
//   loads so staging issues as one clause; one barrier per K-chunk.
// act: 0 none, 1 relu, 2 sigmoid.  accum: out += result (bias still added).
// ---------------------------------------------------------------------------
__global__ __launch_bounds__(256)
void conv_gemm_wmma_fast(const float* __restrict__ in, const unsigned short* __restrict__ wb,
                         const float* __restrict__ bias, float* __restrict__ out,
                         int N, int H, int W, int Cin, int Cout, int KH, int KW,
                         int act, int accum)
{
  __shared__ unsigned int sA[2][128 * 17];   // [buf][m][kpair] bf16x2
  __shared__ unsigned int sB[2][64 * 17];    // [buf][n][kpair] (transposed weights)

  const int tid  = threadIdx.x;
  const int lane = tid & 31;
  const int wv   = tid >> 5;
  const int wm   = wv & 3;            // wave M sub-tile (32 rows each)
  const int wn   = wv >> 2;           // wave N sub-tile (32 cols each)
  const int l16  = lane & 15;
  const int half = lane >> 4;

  const int Mtot  = N * H * W;
  const int mtile = blockIdx.x * 128;
  const int ntile = blockIdx.y * 64;
  const int pad_h = (KH - 1) >> 1, pad_w = (KW - 1) >> 1;

  const int pA  = tid & 15;           // k-pair column owned by this thread
  const int mr0 = tid >> 4;           // base row slot (0..15)

  // ---- hoisted per-slot pixel coordinates (K-invariant) ----
  int  py[8], px[8];
  long pixbase[8];
  bool pv[8];
#pragma unroll
  for (int it = 0; it < 8; ++it) {
    int m  = mtile + mr0 + it * 16;
    bool v = (m < Mtot);
    int mm = v ? m : 0;
    int xw = mm % W; int t = mm / W; int yh = t % H; int nb = t / H;
    py[it] = yh; px[it] = xw; pv[it] = v;
    pixbase[it] = ((long)nb * H + yh) * W + xw;
  }

  // staging registers (chunk being staged)
  float2 fA[8]; bool av[8]; unsigned int fB[4];

  auto load_chunk = [&](int dy, int dx, int c0, int qk) {
#pragma unroll
    for (int it = 0; it < 8; ++it) {
      int iy = py[it] + dy, ix = px[it] + dx;
      bool v = pv[it] && iy >= 0 && iy < H && ix >= 0 && ix < W;
      av[it] = v;
      long off = v ? (pixbase[it] + (long)dy * W + dx) * (long)Cin : 0;
      fA[it] = *(const float2*)(in + off + c0 + 2 * pA);     // unconditional
    }
    long kbase = 32L * qk + 2 * pA;
#pragma unroll
    for (int it = 0; it < 4; ++it) {
      int n  = ntile + mr0 + it * 16;
      int nc = (n < Cout) ? n : (Cout - 1);                  // clamped
      fB[it] = (unsigned)wb[kbase * Cout + nc]
             | ((unsigned)wb[(kbase + 1) * Cout + nc] << 16);
    }
  };
  auto store_chunk = [&](int b) {
#pragma unroll
    for (int it = 0; it < 8; ++it)
      sA[b][(mr0 + it * 16) * 17 + pA] = av[it] ? pack_bf16(fA[it].x, fA[it].y) : 0u;
#pragma unroll
    for (int it = 0; it < 4; ++it) {
      int n = ntile + mr0 + it * 16;
      sB[b][(mr0 + it * 16) * 17 + pA] = (n < Cout) ? fB[it] : 0u;
    }
  };

  v8f c00 = {0.f,0.f,0.f,0.f,0.f,0.f,0.f,0.f};
  v8f c01 = c00, c10 = c00, c11 = c00;

  const int NC = KH * KW * (Cin >> 5);

  int scy = 0, scx = 0, sc0 = 0;      // staging tap params
  auto advance = [&]() {
    sc0 += 32;
    if (sc0 == Cin) { sc0 = 0; ++scx; if (scx == KW) { scx = 0; ++scy; } }
  };

  load_chunk(scy - pad_h, scx - pad_w, sc0, 0);
  store_chunk(0);

  for (int q = 0; q < NC; ++q) {
    const int  cur  = q & 1;
    const bool more = (q + 1 < NC);
    if (more) { advance(); load_chunk(scy - pad_h, scx - pad_w, sc0, q + 1); }
    __syncthreads();                              // LDS[cur] ready

    FragBits a0, a1, b0, b1;
    const int ar0 = (wm * 32 + l16) * 17, ar1 = (wm * 32 + 16 + l16) * 17;
    const int br0 = (wn * 32 + l16) * 17, br1 = (wn * 32 + 16 + l16) * 17;
#pragma unroll
    for (int v = 0; v < 8; ++v) {
      int p = (v & 3) + (half << 2) + ((v >> 2) << 3);   // K-pair index
      a0.u[v] = sA[cur][ar0 + p];
      a1.u[v] = sA[cur][ar1 + p];
      b0.u[v] = sB[cur][br0 + p];
      b1.u[v] = sB[cur][br1 + p];
    }
    v16bf A0 = __builtin_bit_cast(v16bf, a0);
    v16bf A1 = __builtin_bit_cast(v16bf, a1);
    v16bf B0 = __builtin_bit_cast(v16bf, b0);
    v16bf B1 = __builtin_bit_cast(v16bf, b1);
    c00 = wmma_bf16(A0, B0, c00);
    c01 = wmma_bf16(A0, B1, c01);
    c10 = wmma_bf16(A1, B0, c10);
    c11 = wmma_bf16(A1, B1, c11);

    if (more) store_chunk(1 - cur);               // fill the other buffer
  }

#pragma unroll
  for (int sm = 0; sm < 2; ++sm) {
#pragma unroll
    for (int sn = 0; sn < 2; ++sn) {
      v8f c = sm ? (sn ? c11 : c10) : (sn ? c01 : c00);
      int n = ntile + wn * 32 + sn * 16 + l16;
      if (n >= Cout) continue;
      float bs = bias ? bias[n] : 0.f;
#pragma unroll
      for (int r = 0; r < 8; ++r) {
        int m = mtile + wm * 32 + sm * 16 + r + half * 8;
        if (m >= Mtot) continue;
        float v = c[r] + bs;
        long o = (long)m * Cout + n;
        if (accum) v += out[o];
        out[o] = apply_act(v, act);
      }
    }
  }
}

// ---------------------------------------------------------------------------
// WMMA conv_transpose 4x4 stride 2 (SAME, out = 2x in), Cin % 32 == 0.
//   Decomposed into 4 output-parity classes (blockIdx.z = py*2+px); each
//   parity is a stride-1 implicit GEMM over the INPUT grid with 2x2 taps:
//   K = 4*Cin.  Same double-buffered, branchless pipeline as the fast conv.
//   Output written with bias, no activation (cv1 accumulates on top).
// ---------------------------------------------------------------------------
__global__ __launch_bounds__(256)
void convT_gemm_wmma(const float* __restrict__ in, const unsigned short* __restrict__ wb,
                     const float* __restrict__ bias, float* __restrict__ out,
                     int N, int H, int W, int Cin, int Cout)
{
  __shared__ unsigned int sA[2][128 * 17];
  __shared__ unsigned int sB[2][64 * 17];

  const int tid  = threadIdx.x;
  const int lane = tid & 31;
  const int wv   = tid >> 5;
  const int wm   = wv & 3;
  const int wn   = wv >> 2;
  const int l16  = lane & 15;
  const int half = lane >> 4;

  const int Mtot  = N * H * W;          // input-grid pixels
  const int mtile = blockIdx.x * 128;
  const int ntile = blockIdx.y * 64;
  const int par   = blockIdx.z;         // output parity class
  const int pary  = par >> 1, parx = par & 1;
  const int OH = 2 * H, OW = 2 * W;

  const int pA  = tid & 15;
  const int mr0 = tid >> 4;

  int  py[8], px[8];
  long pixbase[8];
  bool pv[8];
#pragma unroll
  for (int it = 0; it < 8; ++it) {
    int m  = mtile + mr0 + it * 16;
    bool v = (m < Mtot);
    int mm = v ? m : 0;
    int xw = mm % W; int t = mm / W; int yh = t % H; int nb = t / H;
    py[it] = yh; px[it] = xw; pv[it] = v;
    pixbase[it] = ((long)nb * H + yh) * W + xw;
  }

  float2 fA[8]; bool av[8]; unsigned int fB[4];

  auto load_chunk = [&](int dy, int dx, long kb, int c0) {
#pragma unroll
    for (int it = 0; it < 8; ++it) {
      int iy = py[it] + dy, ix = px[it] + dx;
      bool v = pv[it] && iy >= 0 && iy < H && ix >= 0 && ix < W;
      av[it] = v;
      long off = v ? (pixbase[it] + (long)dy * W + dx) * (long)Cin : 0;
      fA[it] = *(const float2*)(in + off + c0 + 2 * pA);
    }
    long kk = kb + c0 + 2 * pA;
#pragma unroll
    for (int it = 0; it < 4; ++it) {
      int n  = ntile + mr0 + it * 16;
      int nc = (n < Cout) ? n : (Cout - 1);
      fB[it] = (unsigned)wb[kk * Cout + nc]
             | ((unsigned)wb[(kk + 1) * Cout + nc] << 16);
    }
  };
  auto store_chunk = [&](int b) {
#pragma unroll
    for (int it = 0; it < 8; ++it)
      sA[b][(mr0 + it * 16) * 17 + pA] = av[it] ? pack_bf16(fA[it].x, fA[it].y) : 0u;
#pragma unroll
    for (int it = 0; it < 4; ++it) {
      int n = ntile + mr0 + it * 16;
      sB[b][(mr0 + it * 16) * 17 + pA] = (n < Cout) ? fB[it] : 0u;
    }
  };

  v8f c00 = {0.f,0.f,0.f,0.f,0.f,0.f,0.f,0.f};
  v8f c01 = c00, c10 = c00, c11 = c00;

  const int NC = 4 * (Cin >> 5);        // 2x2 taps * Cin/32 chunks
  int tty = 0, ttx = 0, tc0 = 0;        // staging tap params
  auto advance = [&]() {
    tc0 += 32;
    if (tc0 == Cin) { tc0 = 0; ++ttx; if (ttx == 2) { ttx = 0; ++tty; } }
  };
  // tap tables (uniform scalar selects):
  //   oy = 2y+pary: valid ky have (oy+1-ky) even; iy = y + dy
  //   pary=0: (ky=1,dy=0),(ky=3,dy=-1)   pary=1: (ky=2,dy=0),(ky=0,dy=+1)
  auto tap_dy = [&](int tt) { return tt ? (pary ? 1 : -1) : 0; };
  auto tap_ky = [&](int tt) { return tt ? (pary ? 0 : 3) : (pary ? 2 : 1); };
  auto tap_dx = [&](int tt) { return tt ? (parx ? 1 : -1) : 0; };
  auto tap_kx = [&](int tt) { return tt ? (parx ? 0 : 3) : (parx ? 2 : 1); };

  load_chunk(tap_dy(0), tap_dx(0), (long)(tap_ky(0) * 4 + tap_kx(0)) * Cin, 0);
  store_chunk(0);

  for (int q = 0; q < NC; ++q) {
    const int  cur  = q & 1;
    const bool more = (q + 1 < NC);
    if (more) {
      advance();
      load_chunk(tap_dy(tty), tap_dx(ttx),
                 (long)(tap_ky(tty) * 4 + tap_kx(ttx)) * Cin, tc0);
    }
    __syncthreads();

    FragBits a0, a1, b0, b1;
    const int ar0 = (wm * 32 + l16) * 17, ar1 = (wm * 32 + 16 + l16) * 17;
    const int br0 = (wn * 32 + l16) * 17, br1 = (wn * 32 + 16 + l16) * 17;
#pragma unroll
    for (int v = 0; v < 8; ++v) {
      int p = (v & 3) + (half << 2) + ((v >> 2) << 3);
      a0.u[v] = sA[cur][ar0 + p];
      a1.u[v] = sA[cur][ar1 + p];
      b0.u[v] = sB[cur][br0 + p];
      b1.u[v] = sB[cur][br1 + p];
    }
    v16bf A0 = __builtin_bit_cast(v16bf, a0);
    v16bf A1 = __builtin_bit_cast(v16bf, a1);
    v16bf B0 = __builtin_bit_cast(v16bf, b0);
    v16bf B1 = __builtin_bit_cast(v16bf, b1);
    c00 = wmma_bf16(A0, B0, c00);
    c01 = wmma_bf16(A0, B1, c01);
    c10 = wmma_bf16(A1, B0, c10);
    c11 = wmma_bf16(A1, B1, c11);

    if (more) store_chunk(1 - cur);
  }

  // epilogue: scatter to parity-strided output pixels
#pragma unroll
  for (int sm = 0; sm < 2; ++sm) {
#pragma unroll
    for (int sn = 0; sn < 2; ++sn) {
      v8f c = sm ? (sn ? c11 : c10) : (sn ? c01 : c00);
      int n = ntile + wn * 32 + sn * 16 + l16;
      if (n >= Cout) continue;
      float bs = bias ? bias[n] : 0.f;
#pragma unroll
      for (int r = 0; r < 8; ++r) {
        int m = mtile + wm * 32 + sm * 16 + r + half * 8;
        if (m >= Mtot) continue;
        int xw = m % W; int t = m / W; int yh = t % H; int nb = t / H;
        long o = (((long)nb * OH + 2 * yh + pary) * OW + 2 * xw + parx) * Cout + n;
        out[o] = c[r] + bs;
      }
    }
  }
}

// ---------------------------------------------------------------------------
// GENERIC implicit-GEMM conv (any Cin / K; used only for Cin=3 and Cin=1).
// ---------------------------------------------------------------------------
__global__ __launch_bounds__(256)
void conv_gemm_wmma_gen(const float* __restrict__ in, const unsigned short* __restrict__ wb,
                        const float* __restrict__ bias, float* __restrict__ out,
                        int N, int H, int W, int Cin, int Cout, int KH, int KW,
                        int act, int accum)
{
  __shared__ unsigned int sA[64 * 17];
  __shared__ unsigned int sB[64 * 17];

  const int tid  = threadIdx.x;
  const int lane = tid & 31;
  const int wv   = tid >> 5;
  const int wm   = wv & 3;
  const int wn   = wv >> 2;
  const int l16  = lane & 15;
  const int half = lane >> 4;

  const int Mtot  = N * H * W;
  const int mtile = blockIdx.x * 64;
  const int ntile = blockIdx.y * 64;
  const int Ktot  = KH * KW * Cin;
  const int pad_h = (KH - 1) >> 1, pad_w = (KW - 1) >> 1;

  v8f acc0 = {0.f,0.f,0.f,0.f,0.f,0.f,0.f,0.f};
  v8f acc1 = acc0;

  for (int k0 = 0; k0 < Ktot; k0 += 32) {
#pragma unroll
    for (int it = 0; it < 4; ++it) {
      int idx = tid + it * 256;
      int mr  = idx >> 4;
      int p   = idx & 15;
      int m   = mtile + mr;
      unsigned int packed = 0u;
      if (m < Mtot) {
        int xw = m % W; int t = m / W; int yh = t % H; int nb = t / H;
        float f0 = 0.f, f1 = 0.f;
        int kg = k0 + (p << 1);
        if (kg < Ktot) {
          int ci = kg % Cin; int kk = kg / Cin;
          int kx = kk % KW;  int ky = kk / KW;
          int iy = yh + ky - pad_h, ix = xw + kx - pad_w;
          if (iy >= 0 && iy < H && ix >= 0 && ix < W)
            f0 = in[(((long)nb * H + iy) * W + ix) * Cin + ci];
        }
        if (kg + 1 < Ktot) {
          int kg1 = kg + 1;
          int ci = kg1 % Cin; int kk = kg1 / Cin;
          int kx = kk % KW;   int ky = kk / KW;
          int iy = yh + ky - pad_h, ix = xw + kx - pad_w;
          if (iy >= 0 && iy < H && ix >= 0 && ix < W)
            f1 = in[(((long)nb * H + iy) * W + ix) * Cin + ci];
        }
        packed = pack_bf16(f0, f1);
      }
      sA[mr * 17 + p] = packed;
    }
#pragma unroll
    for (int it = 0; it < 4; ++it) {
      int idx = tid + it * 256;
      int nr  = idx >> 4;
      int p   = idx & 15;
      int n   = ntile + nr;
      unsigned int packed = 0u;
      if (n < Cout) {
        int kg = k0 + (p << 1);
        unsigned int b0 = (kg     < Ktot) ? (unsigned int)wb[(long)kg * Cout + n]       : 0u;
        unsigned int b1 = (kg + 1 < Ktot) ? (unsigned int)wb[((long)kg + 1) * Cout + n] : 0u;
        packed = b0 | (b1 << 16);
      }
      sB[nr * 17 + p] = packed;
    }
    __syncthreads();

    FragBits au, bu0, bu1;
    const int arow  = (wm * 16 + l16) * 17;
    const int brow0 = (wn * 32 + l16) * 17;
    const int brow1 = (wn * 32 + 16 + l16) * 17;
#pragma unroll
    for (int v = 0; v < 8; ++v) {
      int p = (v & 3) + (half << 2) + ((v >> 2) << 3);
      au.u[v]  = sA[arow  + p];
      bu0.u[v] = sB[brow0 + p];
      bu1.u[v] = sB[brow1 + p];
    }
    v16bf av  = __builtin_bit_cast(v16bf, au);
    v16bf bv0 = __builtin_bit_cast(v16bf, bu0);
    v16bf bv1 = __builtin_bit_cast(v16bf, bu1);
    acc0 = wmma_bf16(av, bv0, acc0);
    acc1 = wmma_bf16(av, bv1, acc1);
    __syncthreads();
  }

#pragma unroll
  for (int sub = 0; sub < 2; ++sub) {
    v8f c = sub ? acc1 : acc0;
    int n = ntile + wn * 32 + sub * 16 + l16;
    if (n >= Cout) continue;
    float bs = bias ? bias[n] : 0.f;
#pragma unroll
    for (int r = 0; r < 8; ++r) {
      int m = mtile + wm * 16 + r + half * 8;
      if (m >= Mtot) continue;
      float v = c[r] + bs;
      long o = (long)m * Cout + n;
      if (accum) v += out[o];
      out[o] = apply_act(v, act);
    }
  }
}

// ---------------------------------------------------------------------------
// 2x2 max pool, stride 2
// ---------------------------------------------------------------------------
__global__ void maxpool2(const float* __restrict__ in, float* __restrict__ out,
                         int N, int H, int W, int C)
{
  int Ho = H >> 1, Wo = W >> 1;
  long idx = (long)blockIdx.x * blockDim.x + threadIdx.x;
  long tot = (long)N * Ho * Wo * C;
  if (idx >= tot) return;
  int c = idx % C; long t = idx / C;
  int ox = t % Wo; t /= Wo; int oy = t % Ho; int nb = (int)(t / Ho);
  long base = (((long)nb * H + 2 * oy) * W + 2 * ox) * C + c;
  float m = in[base];
  m = fmaxf(m, in[base + C]);
  m = fmaxf(m, in[base + (long)W * C]);
  m = fmaxf(m, in[base + (long)W * C + C]);
  out[idx] = m;
}

// ---------------------------------------------------------------------------
// Fixed directional kernels (h/v/diag/antidiag line sums, window d=2r+1),
// fused with the outer a* multiply:  out = a * (sum of 4 line sums of x)
// ---------------------------------------------------------------------------
__global__ void nl_dirsum(const float* __restrict__ a, const float* __restrict__ x,
                          float* __restrict__ out, int N, int H, int W, int r)
{
  long idx = (long)blockIdx.x * blockDim.x + threadIdx.x;
  long tot = (long)N * H * W;
  if (idx >= tot) return;
  int xx = idx % W; long t = idx / W; int y = t % H; int nb = (int)(t / H);
  const float* xp = x + (long)nb * H * W;
  float s = 0.f;
  for (int tt = -r; tt <= r; ++tt) {
    int yy = y + tt;  bool yv = (yy >= 0 && yy < H);
    int xc = xx + tt; bool xv = (xc >= 0 && xc < W);
    if (yv)       s += xp[(long)yy * W + xx];       // vertical (ones(d,1))
    if (xv)       s += xp[(long)y  * W + xc];       // horizontal (ones(1,d))
    if (yv && xv) s += xp[(long)yy * W + xc];       // diagonal (eye)
    int xm = xx - tt;
    if (yv && xm >= 0 && xm < W) s += xp[(long)yy * W + xm]; // anti-diagonal
  }
  out[idx] = a[idx] * s;
}

__global__ void emul(const float* __restrict__ a, const float* __restrict__ b,
                     float* __restrict__ o, long n)
{
  long i = (long)blockIdx.x * blockDim.x + threadIdx.x;
  if (i < n) o[i] = a[i] * b[i];
}

// concat along channels: out[p][0..d) = A, out[p][d..2d) = B
__global__ void concat2(const float* __restrict__ A, const float* __restrict__ B,
                        float* __restrict__ o, long P, int d)
{
  long idx = (long)blockIdx.x * blockDim.x + threadIdx.x;
  long tot = P * 2 * d;
  if (idx >= tot) return;
  int c = idx % (2 * d); long p = idx / (2 * d);
  o[idx] = (c < d) ? A[p * d + c] : B[p * d + (c - d)];
}

// bilinear resize SxS -> 512x512 (half-pixel centers, like jax.image.resize)
__global__ void resize_bil(const float* __restrict__ in, float* __restrict__ out,
                           int N, int S, int C)
{
  long idx = (long)blockIdx.x * blockDim.x + threadIdx.x;
  long tot = (long)N * 512 * 512 * C;
  if (idx >= tot) return;
  int c = idx % C; long t = idx / C;
  int ox = t % 512; t /= 512; int oy = t % 512; int nb = (int)(t / 512);
  float sc = (float)S * (1.0f / 512.0f);
  float fy = (oy + 0.5f) * sc - 0.5f;
  float fx = (ox + 0.5f) * sc - 0.5f;
  int y0 = (int)floorf(fy); float wy = fy - (float)y0;
  int x0 = (int)floorf(fx); float wx = fx - (float)x0;
  int y1 = y0 + 1, x1 = x0 + 1;
  y0 = min(max(y0, 0), S - 1); y1 = min(max(y1, 0), S - 1);
  x0 = min(max(x0, 0), S - 1); x1 = min(max(x1, 0), S - 1);
  const float* ip = in + ((long)nb * S) * S * C + c;
  float v00 = ip[((long)y0 * S + x0) * C], v01 = ip[((long)y0 * S + x1) * C];
  float v10 = ip[((long)y1 * S + x0) * C], v11 = ip[((long)y1 * S + x1) * C];
  out[idx] = v00 * (1.f - wy) * (1.f - wx) + v01 * (1.f - wy) * wx
           + v10 * wy * (1.f - wx)        + v11 * wy * wx;
}

// ---------------------------------------------------------------------------
// Host orchestration
// ---------------------------------------------------------------------------
static inline int idiv(long a, int b) { return (int)((a + b - 1) / b); }

struct Conv {
  const float* w; const float* b; unsigned short* wb;
  int kh, kw, ci, co;
};

extern "C" void kernel_launch(void* const* d_in, const int* in_sizes, int n_in,
                              void* d_out, int out_size, void* d_ws, size_t ws_size,
                              hipStream_t stream)
{
  (void)in_sizes; (void)n_in; (void)out_size; (void)ws_size;

  // ---- gather inputs in setup_inputs() insertion order ----
  int ii = 0;
  const float* X = (const float*)d_in[ii++];           // (2,512,512,3)
  auto take = [&](int kh, int kw, int ci, int co) -> Conv {
    Conv c; c.w = (const float*)d_in[ii++]; c.b = (const float*)d_in[ii++];
    c.kh = kh; c.kw = kw; c.ci = ci; c.co = co; c.wb = nullptr; return c;
  };

  const int DIMS[4] = {256, 128, 64, 32};
  const int SKIP[4] = {512, 256, 128, 64};
  const int DAK[4]  = {9, 17, 33, 65};

  Conv vgg[13];
  {
    const int co[13] = {64, 64, 128, 128, 256, 256, 256, 512, 512, 512, 512, 512, 512};
    int ci = 3;
    for (int i = 0; i < 13; ++i) { vgg[i] = take(3, 3, ci, co[i]); ci = co[i]; }
  }
  struct Branch { Conv ups[4], cv1[4], cv2[4], fin; };
  auto takeBranch = [&](int fo) -> Branch {
    Branch b; int cprev = 512;
    for (int i = 0; i < 4; ++i) { b.ups[i] = take(4, 4, cprev, DIMS[i]); cprev = DIMS[i]; }
    for (int i = 0; i < 4; ++i) b.cv1[i] = take(3, 3, SKIP[i], DIMS[i]);
    for (int i = 0; i < 4; ++i) b.cv2[i] = take(3, 3, DIMS[i], DIMS[i]);
    b.fin = take(1, 1, 32, fo);
    return b;
  };
  Branch rbp = takeBranch(3);
  Branch rtp = takeBranch(9);
  Conv atts1[4], atts2[4], atts3[4], xs1[4], xs2[4], ed[4], lrf[4];
  for (int i = 0; i < 4; ++i) atts1[i] = take(3, 3, DIMS[i], DIMS[i]);
  for (int i = 0; i < 4; ++i) atts2[i] = take(3, 3, DIMS[i], DIMS[i]);
  for (int i = 0; i < 4; ++i) atts3[i] = take(1, 1, DIMS[i], 1);
  for (int i = 0; i < 4; ++i) xs1[i]   = take(3, 3, DIMS[i], DIMS[i]);
  for (int i = 0; i < 4; ++i) xs2[i]   = take(1, 1, DIMS[i], 1);
  for (int i = 0; i < 4; ++i) ed[i]    = take(1, 1, 1, DIMS[i]);
  for (int i = 0; i < 4; ++i) lrf[i]   = take(3, 3, 2 * DIMS[i], DIMS[i]);

  // ---- workspace bump allocator ----
  char* wsb = (char*)d_ws; size_t wo = 0;
  auto fal = [&](size_t elems) -> float* {
    float* p = (float*)(wsb + wo); wo += ((elems * 4 + 255) / 256) * 256; return p;
  };
  auto hal = [&](size_t elems) -> unsigned short* {
    unsigned short* p = (unsigned short*)(wsb + wo); wo += ((elems * 2 + 255) / 256) * 256; return p;
  };

  // ---- pack all WMMA weights to bf16 ----
  auto prep = [&](Conv& c) {
    size_t e = (size_t)c.kh * c.kw * c.ci * c.co;
    c.wb = hal(e);
    w_to_bf16<<<idiv((long)e, 256), 256, 0, stream>>>(c.w, c.wb, (int)e);
  };
  for (int i = 0; i < 13; ++i) prep(vgg[i]);
  for (int i = 0; i < 4; ++i) {
    prep(rbp.ups[i]); prep(rbp.cv1[i]); prep(rbp.cv2[i]);
    prep(rtp.ups[i]); prep(rtp.cv1[i]); prep(rtp.cv2[i]);
  }
  prep(rbp.fin); prep(rtp.fin);
  for (int i = 0; i < 4; ++i) {
    prep(atts1[i]); prep(atts2[i]); prep(atts3[i]);
    prep(xs1[i]);   prep(xs2[i]);   prep(ed[i]);   prep(lrf[i]);
  }

  auto run_conv = [&](const float* in, const Conv& c, float* out_,
                      int N, int H, int W, int act, int accum) {
    long M = (long)N * H * W;
    if ((c.ci & 31) == 0) {
      dim3 g(idiv(M, 128), idiv(c.co, 64));
      conv_gemm_wmma_fast<<<g, 256, 0, stream>>>(in, c.wb, c.b, out_,
                                                 N, H, W, c.ci, c.co, c.kh, c.kw, act, accum);
    } else {
      dim3 g(idiv(M, 64), idiv(c.co, 64));
      conv_gemm_wmma_gen<<<g, 256, 0, stream>>>(in, c.wb, c.b, out_,
                                                N, H, W, c.ci, c.co, c.kh, c.kw, act, accum);
    }
  };
  auto run_convT = [&](const float* in, const Conv& c, float* out_,
                       int N, int H, int W) {
    long M = (long)N * H * W;
    dim3 g(idiv(M, 128), idiv(c.co, 64), 4);
    convT_gemm_wmma<<<g, 256, 0, stream>>>(in, c.wb, c.b, out_, N, H, W, c.ci, c.co);
  };

  // ---- activation buffers ----
  float* A  = fal(2L * 512 * 512 * 64);
  float* B  = fal(2L * 512 * 512 * 64);
  float* p0 = fal(2L * 256 * 256 * 64);
  float* p1 = fal(2L * 128 * 128 * 128);
  float* p2 = fal(2L * 64 * 64 * 256);
  float* p3 = fal(2L * 32 * 32 * 512);
  float* p4 = fal(2L * 16 * 16 * 512);
  float* feats[4];
  feats[0] = fal(2L * 32 * 32 * 256);
  feats[1] = fal(2L * 64 * 64 * 128);
  feats[2] = fal(2L * 128 * 128 * 64);
  feats[3] = fal(2L * 256 * 256 * 32);
  float* bufU  = fal(2L * 256 * 256 * 32);
  float* bufT1 = fal(2L * 256 * 256 * 32);
  float* bufT2 = fal(2L * 256 * 256 * 32);
  float* bufN1 = fal(2L * 256 * 256 * 32);
  float* bufN2 = fal(2L * 256 * 256 * 32);
  float* bufS1 = fal(2L * 256 * 256);
  float* bufS2 = fal(2L * 256 * 256);
  float* bufS3 = fal(2L * 256 * 256);
  float* bufC  = fal(2L * 256 * 256 * 64);
  float* logitS = fal(2L * 256 * 256 * 9);

  // ---- VGG16 encoder ----
  run_conv(X, vgg[0], A, 2, 512, 512, 1, 0);
  run_conv(A, vgg[1], B, 2, 512, 512, 1, 0);
  maxpool2<<<idiv(2L * 256 * 256 * 64, 256), 256, 0, stream>>>(B, p0, 2, 512, 512, 64);
  run_conv(p0, vgg[2], A, 2, 256, 256, 1, 0);
  run_conv(A,  vgg[3], B, 2, 256, 256, 1, 0);
  maxpool2<<<idiv(2L * 128 * 128 * 128, 256), 256, 0, stream>>>(B, p1, 2, 256, 256, 128);
  run_conv(p1, vgg[4], A, 2, 128, 128, 1, 0);
  run_conv(A,  vgg[5], B, 2, 128, 128, 1, 0);
  run_conv(B,  vgg[6], A, 2, 128, 128, 1, 0);
  maxpool2<<<idiv(2L * 64 * 64 * 256, 256), 256, 0, stream>>>(A, p2, 2, 128, 128, 256);
  run_conv(p2, vgg[7], A, 2, 64, 64, 1, 0);
  run_conv(A,  vgg[8], B, 2, 64, 64, 1, 0);
  run_conv(B,  vgg[9], A, 2, 64, 64, 1, 0);
  maxpool2<<<idiv(2L * 32 * 32 * 512, 256), 256, 0, stream>>>(A, p3, 2, 64, 64, 512);
  run_conv(p3, vgg[10], A, 2, 32, 32, 1, 0);
  run_conv(A,  vgg[11], B, 2, 32, 32, 1, 0);
  run_conv(B,  vgg[12], A, 2, 32, 32, 1, 0);
  maxpool2<<<idiv(2L * 16 * 16 * 512, 256), 256, 0, stream>>>(A, p4, 2, 32, 32, 512);

  const float* skips[4] = {p3, p2, p1, p0};

  // ---- room-boundary branch (rbp): store per-level feats ----
  {
    const float* xb = p4; int h = 16;
    for (int i = 0; i < 4; ++i) {
      int oh = 2 * h;
      run_convT(xb, rbp.ups[i], bufU, 2, h, h);                // convT + bias
      run_conv(skips[i], rbp.cv1[i], bufU, 2, oh, oh, 0, 1);   // += skip conv
      run_conv(bufU, rbp.cv2[i], feats[i], 2, oh, oh, 1, 0);   // relu
      xb = feats[i]; h = oh;
    }
    run_conv(xb, rbp.fin, logitS, 2, 256, 256, 0, 0);          // 3-ch logits @256
    float* outCW = (float*)d_out + 2L * 512 * 512 * 9;
    resize_bil<<<idiv(2L * 512 * 512 * 3, 256), 256, 0, stream>>>(logitS, outCW, 2, 256, 3);
  }

  // ---- room-type branch (rtp) with non-local context ----
  {
    const float* xt = p4; int h = 16;
    for (int i = 0; i < 4; ++i) {
      int d = DIMS[i], oh = 2 * h;
      long P = 2L * oh * oh;
      run_convT(xt, rtp.ups[i], bufU, 2, h, h);
      run_conv(skips[i], rtp.cv1[i], bufU, 2, oh, oh, 0, 1);
      run_conv(bufU, rtp.cv2[i], bufT1, 2, oh, oh, 1, 0);      // t2

      // non-local: t1 = feats[i], t2 = bufT1
      run_conv(feats[i], atts1[i], bufN1, 2, oh, oh, 1, 0);
      run_conv(bufN1,    atts2[i], bufN2, 2, oh, oh, 1, 0);
      run_conv(bufN2,    atts3[i], bufS1, 2, oh, oh, 2, 0);    // a (sigmoid, 1ch)
      run_conv(bufT1,    xs1[i],   bufN1, 2, oh, oh, 1, 0);
      run_conv(bufN1,    xs2[i],   bufS2, 2, oh, oh, 0, 0);    // x (1ch)
      emul<<<idiv(P, 256), 256, 0, stream>>>(bufS1, bufS2, bufS3, P);     // x = a*x
      nl_dirsum<<<idiv(P, 256), 256, 0, stream>>>(bufS1, bufS3, bufS2,
                                                  2, oh, oh, DAK[i] / 2); // a*(4 dirs)
      run_conv(bufS2, ed[i], bufN2, 2, oh, oh, 0, 0);          // c1 (d ch)
      concat2<<<idiv(P * 2 * d, 256), 256, 0, stream>>>(bufT1, bufN2, bufC, P, d);
      run_conv(bufC, lrf[i], bufT2, 2, oh, oh, 1, 0);          // xt_new (relu)

      xt = bufT2; h = oh;
      float* tmp = bufT2; bufT2 = bufN1; bufN1 = tmp;          // rotate to avoid overwrite
    }
    run_conv(xt, rtp.fin, logitS, 2, 256, 256, 0, 0);          // 9-ch logits @256
    float* outR = (float*)d_out;
    resize_bil<<<idiv(2L * 512 * 512 * 9, 256), 256, 0, stream>>>(logitS, outR, 2, 256, 9);
  }
}
